// MultiHeadAttention_2207613190385
// MI455X (gfx1250) — compile-verified, matrix operand order checked by reading, not run on hardware
//
// MI455X (gfx1250) dual cross-attention block.
// - All matmuls on v_wmma_f32_16x16x32_f16 (f16 in, f32 accumulate).
// - GEMM tiles fed by the Tensor Data Mover (tensor_load_to_lds, TENSORcnt),
//   double-buffered; TDM pad_enable reproduces the 40-half padded LDS rows
//   (64B interval, 16B pad) so fragment ds_loads stay bank-conflict free.
// - Flash-style online-softmax attention, one wave per (b,h,16-query tile).
// Workspace (~210 MB) bump-allocated from d_ws.
#include <hip/hip_runtime.h>
#include <hip/hip_bf16.h>
#include <cstdint>

typedef _Float16 half_t;
typedef __attribute__((ext_vector_type(16))) _Float16 v16h;
typedef __attribute__((ext_vector_type(8)))  _Float16 v8h;
typedef __attribute__((ext_vector_type(8)))  float    v8f;
typedef __attribute__((ext_vector_type(4)))  unsigned int v4u;
typedef __attribute__((ext_vector_type(8)))  int v8i;
typedef __attribute__((ext_vector_type(4)))  int v4i;

#define B_  16
#define LV_ 196
#define LS_ 512
#define DM_ 768
#define FV_ 2048
#define FS_ 1024
#define H_  12
#define DK_ 64

// ---------------------------------------------------------------- conversions
__global__ void cvt_f32_f16(const float* __restrict__ x, half_t* __restrict__ y,
                            size_t n) {
  size_t i = (size_t)blockIdx.x * blockDim.x + threadIdx.x;
  size_t stride = (size_t)gridDim.x * blockDim.x;
  for (; i < n; i += stride) y[i] = (half_t)x[i];
}

// ------------------------------------------------------------ TDM tile loader
// 2D tile (tile_w x tile_h halfs, row stride = rs halfs) from global -> LDS,
// with LDS padding of 16B after every 64B (row of 32 halfs -> 40-half stride).
// D# layout per CDNA5 ISA 8.3-8.6. Wave-level op; EXEC ignored; TENSORcnt.
// This toolchain exposes the 6-arg builtin: (v4u, v8i, v4i, v4i, v8i, i32).
__device__ __forceinline__ void tdm_load_tile_2d(unsigned lds_addr,
                                                 const half_t* gptr,
                                                 unsigned rs, unsigned tile_w,
                                                 unsigned tile_h) {
  unsigned long long ga = (unsigned long long)(uintptr_t)gptr;
  v4u g0;
  g0[0] = 1u;                                        // count=1, user descriptor
  g0[1] = lds_addr;                                  // LDS byte address
  g0[2] = (unsigned)(ga & 0xFFFFFFFFull);            // global_addr[31:0]
  g0[3] = (unsigned)((ga >> 32) & 0x01FFFFFFull)     // global_addr[56:32]
        | 0x80000000u;                               // type=2 ("image")
  const unsigned td0 = rs;                           // tensor_dim0 (in-bounds)
  const unsigned td1 = 1u << 20;                     // tensor_dim1 (large)
  v8i g1;
  g1[0] = (int)((1u << 16)      // data_size = 1 -> 2 bytes
              | (1u << 20)      // pad_enable
              | (3u << 22)      // pad_interval code 3 -> every 16 DWORDs (64B)
              | (3u << 25));    // pad_amount  code 3 -> 4 DWORDs (16B)
  g1[1] = (int)((td0 & 0xFFFFu) << 16);              // [15:0] atomic_addr=0
  g1[2] = (int)((td0 >> 16) | ((td1 & 0xFFFFu) << 16));
  g1[3] = (int)((td1 >> 16) | (tile_w << 16));       // tile_dim0
  g1[4] = (int)(tile_h & 0xFFFFu);                   // tile_dim1 (tile_dim2=0)
  g1[5] = (int)rs;                                   // tensor_dim0_stride[31:0]
  g1[6] = 0;                                         // stride0 hi / stride1 lo
  g1[7] = 0;
  const v4i gz4 = {0, 0, 0, 0};                      // groups 2/3: 2-D tensor
  const v8i gz8 = {0, 0, 0, 0, 0, 0, 0, 0};          // extra group: unused
  __builtin_amdgcn_tensor_load_to_lds(g0, g1, gz4, gz4, gz8, 0);
}

// ---------------------------------------------------------------- GEMM
// C[M,N] (f32, optional) and C16[M,N] (f16, optional) = A[M,K]*W[N,K]^T + bias.
// A, W f16 row-major. M,N multiples of 64; K multiple of 32.
// 256 threads = 8 waves; 64x64 tile; wave computes 16x32 via 2 WMMAs/K-step.
// Tiles arrive via double-buffered TDM descriptors issued by wave 0.
__global__ __launch_bounds__(256) void gemm_f16w(
    const half_t* __restrict__ A, const half_t* __restrict__ W,
    const float* __restrict__ bias, float* __restrict__ C,
    half_t* __restrict__ C16, int M, int N, int K) {
  __shared__ half_t As[2][64 * 40 + 8];   // +8 halfs: trailing TDM pad slack
  __shared__ half_t Ws[2][64 * 40 + 8];
  const int tid  = threadIdx.x;
  const int lane = tid & 31, wave = tid >> 5;
  const int hl = lane >> 4, l15 = lane & 15;
  const int bm = blockIdx.x * 64, bn = blockIdx.y * 64;
  const int wm = (wave & 3) * 16, wn = (wave >> 2) * 32;
  const half_t* Abase = A + (size_t)bm * K;
  const half_t* Wbase = W + (size_t)bn * K;
  v8f acc0 = {}, acc1 = {};
  const int nsteps = K / 32;

  if (tid < 32) {  // wave 0: prologue TDM for buffer 0
    tdm_load_tile_2d((unsigned)(uintptr_t)&As[0][0], Abase, (unsigned)K, 32, 64);
    tdm_load_tile_2d((unsigned)(uintptr_t)&Ws[0][0], Wbase, (unsigned)K, 32, 64);
  }
  for (int step = 0; step < nsteps; ++step) {
    const int cur = step & 1, nxt = cur ^ 1;
    if (tid < 32) {
      if (step + 1 < nsteps) {  // prefetch next tile, then wait for current
        const int kn = (step + 1) * 32;
        tdm_load_tile_2d((unsigned)(uintptr_t)&As[nxt][0], Abase + kn,
                         (unsigned)K, 32, 64);
        tdm_load_tile_2d((unsigned)(uintptr_t)&Ws[nxt][0], Wbase + kn,
                         (unsigned)K, 32, 64);
        __builtin_amdgcn_s_wait_tensorcnt((short)2);
      } else {
        __builtin_amdgcn_s_wait_tensorcnt((short)0);
      }
    }
    __syncthreads();  // current buffer visible to all waves
    v16h a, b0, b1;
    ((v8h*)&a)[0]  = *(const v8h*)&As[cur][(wm + l15) * 40 + hl * 8];
    ((v8h*)&a)[1]  = *(const v8h*)&As[cur][(wm + l15) * 40 + 16 + hl * 8];
    ((v8h*)&b0)[0] = *(const v8h*)&Ws[cur][(wn + l15) * 40 + hl * 8];
    ((v8h*)&b0)[1] = *(const v8h*)&Ws[cur][(wn + l15) * 40 + 16 + hl * 8];
    ((v8h*)&b1)[0] = *(const v8h*)&Ws[cur][(wn + 16 + l15) * 40 + hl * 8];
    ((v8h*)&b1)[1] = *(const v8h*)&Ws[cur][(wn + 16 + l15) * 40 + 16 + hl * 8];
    acc0 = __builtin_amdgcn_wmma_f32_16x16x32_f16(false, a, false, b0, (short)0, acc0, false, false);
    acc1 = __builtin_amdgcn_wmma_f32_16x16x32_f16(false, a, false, b1, (short)0, acc1, false, false);
    __syncthreads();  // all reads done before wave0 overwrites this buffer
  }
  const int n0 = bn + wn + l15, n1 = n0 + 16;
  const float bs0 = bias ? bias[n0] : 0.f;
  const float bs1 = bias ? bias[n1] : 0.f;
#pragma unroll
  for (int j = 0; j < 8; ++j) {
    const int m = bm + wm + hl * 8 + j;
    const float c0 = acc0[j] + bs0, c1 = acc1[j] + bs1;
    if (C)   { C[(size_t)m * N + n0] = c0;          C[(size_t)m * N + n1] = c1; }
    if (C16) { C16[(size_t)m * N + n0] = (half_t)c0; C16[(size_t)m * N + n1] = (half_t)c1; }
  }
}

// ------------------------------------------------- V pre-swizzle (B-fragment order)
// Out layout per (b,h): [LkPad/32 kchunks][4 dim-frags][32 lanes][16 halfs],
// so attention reads one contiguous 32B vector per lane per fragment.
__global__ void vswz_kernel(const half_t* __restrict__ V, half_t* __restrict__ out,
                            int Lk, int LkPad) {
  const int nkc = LkPad >> 5;
  long t = (long)blockIdx.x * blockDim.x + threadIdx.x;
  const long total = (long)B_ * H_ * nkc * 4 * 32;
  if (t >= total) return;
  const int lane = (int)(t & 31); long u = t >> 5;
  const int f = (int)(u & 3); u >>= 2;
  const int kc = (int)(u % nkc); u /= nkc;
  const int h = (int)(u % H_);
  const int b = (int)(u / H_);
  const int hl = lane >> 4, l15 = lane & 15;
  half_t vals[16];
#pragma unroll
  for (int e = 0; e < 16; ++e) {
    int k = kc * 32 + ((e >> 3) << 4) + (hl << 3) + (e & 7);
    if (k >= Lk) k = Lk - 1;  // finite value; masked to P=0 in attention
    vals[e] = V[((size_t)b * Lk + k) * DM_ + h * DK_ + f * 16 + l15];
  }
  half_t* op = out + ((((size_t)(b * H_ + h) * nkc + kc) * 4 + f) * 32 + lane) * 16;
#pragma unroll
  for (int e = 0; e < 16; ++e) op[e] = vals[e];
}

// ---------------------------------------------------------------- attention
// One wave per (b, h, 16-query tile). Flash-style online softmax over Lk in
// 32-key chunks. Q/K fragment loads come straight from the row-major f16
// projection outputs (their layouts match the WMMA A/B per-lane layouts).
__global__ __launch_bounds__(32) void attn_kernel(
    const half_t* __restrict__ Q, const half_t* __restrict__ Km,
    const half_t* __restrict__ Vswz, half_t* __restrict__ ctxo,
    int Lq, int Lk, int LkPad, float scale) {
  const int qtiles = (Lq + 15) >> 4;
  int idx = blockIdx.x;
  const int qt = idx % qtiles; idx /= qtiles;
  const int h = idx % H_;
  const int b = idx / H_;
  const int lane = threadIdx.x & 31;
  const int hl = lane >> 4, l15 = lane & 15;
  __shared__ half_t Ps[16 * 40];

  const int q0 = qt * 16;
  int qrow = q0 + l15; if (qrow >= Lq) qrow = Lq - 1;  // dup rows; store guarded
  const half_t* qp = Q + ((size_t)b * Lq + qrow) * DM_ + h * DK_;
  v16h a0, a1;  // Q fragments for DK chunks [0,32) and [32,64)
  ((v8h*)&a0)[0] = *(const v8h*)(qp + hl * 8);
  ((v8h*)&a0)[1] = *(const v8h*)(qp + 16 + hl * 8);
  ((v8h*)&a1)[0] = *(const v8h*)(qp + 32 + hl * 8);
  ((v8h*)&a1)[1] = *(const v8h*)(qp + 48 + hl * 8);

  float mrun[8], lrun[8];
  v8f ctx0 = {}, ctx1 = {}, ctx2 = {}, ctx3 = {};
#pragma unroll
  for (int j = 0; j < 8; ++j) { mrun[j] = -3.0e30f; lrun[j] = 0.f; }

  const int nkc = LkPad >> 5;
  const half_t* vb_bh = Vswz + ((size_t)(b * H_ + h) * nkc) * (4 * 32 * 16);

  for (int k0 = 0; k0 < LkPad; k0 += 32) {
    // ---- scores S[16 q x 32 k] = Q * K^T ----
    v8f s0 = {}, s1 = {};
    const int key0 = k0 + l15, key1 = k0 + 16 + l15;
    const int key0c = key0 < Lk ? key0 : Lk - 1;
    const int key1c = key1 < Lk ? key1 : Lk - 1;
    const half_t* kp0 = Km + ((size_t)b * Lk + key0c) * DM_ + h * DK_;
    const half_t* kp1 = Km + ((size_t)b * Lk + key1c) * DM_ + h * DK_;
    v16h b00, b01, b10, b11;
    ((v8h*)&b00)[0] = *(const v8h*)(kp0 + hl * 8);
    ((v8h*)&b00)[1] = *(const v8h*)(kp0 + 16 + hl * 8);
    ((v8h*)&b01)[0] = *(const v8h*)(kp0 + 32 + hl * 8);
    ((v8h*)&b01)[1] = *(const v8h*)(kp0 + 48 + hl * 8);
    ((v8h*)&b10)[0] = *(const v8h*)(kp1 + hl * 8);
    ((v8h*)&b10)[1] = *(const v8h*)(kp1 + 16 + hl * 8);
    ((v8h*)&b11)[0] = *(const v8h*)(kp1 + 32 + hl * 8);
    ((v8h*)&b11)[1] = *(const v8h*)(kp1 + 48 + hl * 8);
    s0 = __builtin_amdgcn_wmma_f32_16x16x32_f16(false, a0, false, b00, (short)0, s0, false, false);
    s0 = __builtin_amdgcn_wmma_f32_16x16x32_f16(false, a1, false, b01, (short)0, s0, false, false);
    s1 = __builtin_amdgcn_wmma_f32_16x16x32_f16(false, a0, false, b10, (short)0, s1, false, false);
    s1 = __builtin_amdgcn_wmma_f32_16x16x32_f16(false, a1, false, b11, (short)0, s1, false, false);

    // ---- online softmax across this 32-key chunk ----
    const bool dead0 = key0 >= Lk, dead1 = key1 >= Lk;
#pragma unroll
    for (int j = 0; j < 8; ++j) {
      float x0 = dead0 ? -3.0e30f : s0[j] * scale;
      float x1 = dead1 ? -3.0e30f : s1[j] * scale;
      float mx = fmaxf(x0, x1);
#pragma unroll
      for (int o = 1; o < 16; o <<= 1) mx = fmaxf(mx, __shfl_xor(mx, o, 32));
      const float mnew = fmaxf(mrun[j], mx);
      const float corr = __expf(mrun[j] - mnew);
      mrun[j] = mnew;
      const float e0 = __expf(x0 - mnew), e1 = __expf(x1 - mnew);
      float rs = e0 + e1;
#pragma unroll
      for (int o = 1; o < 16; o <<= 1) rs += __shfl_xor(rs, o, 32);
      lrun[j] = lrun[j] * corr + rs;
      ctx0[j] *= corr; ctx1[j] *= corr; ctx2[j] *= corr; ctx3[j] *= corr;
      // stage P tile (C-layout -> LDS row-major), row = query, col = key
      Ps[(hl * 8 + j) * 40 + l15]      = (half_t)e0;
      Ps[(hl * 8 + j) * 40 + 16 + l15] = (half_t)e1;
    }
    asm volatile("s_wait_dscnt 0" ::: "memory");  // single wave: order LDS st->ld
    v16h pA;  // P tile re-read in WMMA A-layout
    ((v8h*)&pA)[0] = *(const v8h*)&Ps[l15 * 40 + hl * 8];
    ((v8h*)&pA)[1] = *(const v8h*)&Ps[l15 * 40 + 16 + hl * 8];

    // ---- ctx += P * V (pre-swizzled) ----
    const half_t* vb = vb_bh + (size_t)(k0 >> 5) * (4 * 32 * 16);
    v16h vf0 = *(const v16h*)(vb + (0 * 32 + lane) * 16);
    v16h vf1 = *(const v16h*)(vb + (1 * 32 + lane) * 16);
    v16h vf2 = *(const v16h*)(vb + (2 * 32 + lane) * 16);
    v16h vf3 = *(const v16h*)(vb + (3 * 32 + lane) * 16);
    ctx0 = __builtin_amdgcn_wmma_f32_16x16x32_f16(false, pA, false, vf0, (short)0, ctx0, false, false);
    ctx1 = __builtin_amdgcn_wmma_f32_16x16x32_f16(false, pA, false, vf1, (short)0, ctx1, false, false);
    ctx2 = __builtin_amdgcn_wmma_f32_16x16x32_f16(false, pA, false, vf2, (short)0, ctx2, false, false);
    ctx3 = __builtin_amdgcn_wmma_f32_16x16x32_f16(false, pA, false, vf3, (short)0, ctx3, false, false);
  }

#pragma unroll
  for (int j = 0; j < 8; ++j) {
    const int m = q0 + hl * 8 + j;
    if (m < Lq) {
      const float inv = 1.f / lrun[j];
      half_t* op = ctxo + ((size_t)b * Lq + m) * DM_ + h * DK_;
      op[l15]      = (half_t)(ctx0[j] * inv);
      op[16 + l15] = (half_t)(ctx1[j] * inv);
      op[32 + l15] = (half_t)(ctx2[j] * inv);
      op[48 + l15] = (half_t)(ctx3[j] * inv);
    }
  }
}

// --------------------------------------- residual + pos-emb + LayerNorm (ddof=1)
__global__ __launch_bounds__(256) void ln_fuse(
    const float* __restrict__ projV, const float* __restrict__ projS,
    const float* __restrict__ v2, const float* __restrict__ s2,
    const int* __restrict__ vlen, const int* __restrict__ slen,
    const float* __restrict__ posV, const float* __restrict__ posS,
    const float* __restrict__ gv, const float* __restrict__ bev,
    const float* __restrict__ gs, const float* __restrict__ bes,
    float* __restrict__ out) {
  __shared__ float red[256];
  const int r = blockIdx.x;
  const float *proj, *res, *pos, *g, *be;
  const int* lens;
  int L, row;
  size_t obase;
  if (r < B_ * LV_) {
    proj = projV; res = v2; pos = posV; g = gv; be = bev;
    lens = vlen; L = LV_; row = r; obase = (size_t)r * DM_;
  } else {
    const int r2 = r - B_ * LV_;
    proj = projS; res = s2; pos = posS; g = gs; be = bes;
    lens = slen; L = LS_; row = r2;
    obase = (size_t)B_ * LV_ * DM_ + (size_t)r2 * DM_;
  }
  const int b = row / L, p = row % L;
  const int pidx = (p + 1 <= lens[b]) ? (p + 1) : 0;
  const float* prow  = proj + (size_t)row * DM_;
  const float* rrow  = res  + (size_t)row * DM_;
  const float* porow = pos  + (size_t)pidx * DM_;
  const int t = threadIdx.x;
  float z[3];
  float s = 0.f;
#pragma unroll
  for (int q = 0; q < 3; ++q) {
    const int i = t + q * 256;
    z[q] = prow[i] + rrow[i] + porow[i];
    s += z[q];
  }
  red[t] = s; __syncthreads();
  for (int o = 128; o > 0; o >>= 1) { if (t < o) red[t] += red[t + o]; __syncthreads(); }
  const float mean = red[0] / DM_;
  __syncthreads();
  float ss = 0.f;
#pragma unroll
  for (int q = 0; q < 3; ++q) { const float d = z[q] - mean; ss += d * d; }
  red[t] = ss; __syncthreads();
  for (int o = 128; o > 0; o >>= 1) { if (t < o) red[t] += red[t + o]; __syncthreads(); }
  const float sd = sqrtf(red[0] / (float)(DM_ - 1));
  const float inv = 1.f / (sd + 1e-6f);
#pragma unroll
  for (int q = 0; q < 3; ++q) {
    const int i = t + q * 256;
    out[obase + i] = g[i] * (z[q] - mean) * inv + be[i];
  }
}

// ---------------------------------------------------------------- launcher
extern "C" void kernel_launch(void* const* d_in, const int* in_sizes, int n_in,
                              void* d_out, int out_size, void* d_ws, size_t ws_size,
                              hipStream_t stream) {
  (void)in_sizes; (void)n_in; (void)out_size; (void)ws_size;
  const float* v     = (const float*)d_in[0];
  const float* s     = (const float*)d_in[1];
  const int*   v_len = (const int*)d_in[2];
  const int*   s_len = (const int*)d_in[3];
  const float* Wlv = (const float*)d_in[4];   const float* blv = (const float*)d_in[5];
  const float* Wls = (const float*)d_in[6];   const float* bls = (const float*)d_in[7];
  const float* Wq_v = (const float*)d_in[8];  const float* bq_v = (const float*)d_in[9];
  const float* Wk_v = (const float*)d_in[10]; const float* bk_v = (const float*)d_in[11];
  const float* Wv_v = (const float*)d_in[12]; const float* bv_v = (const float*)d_in[13];
  const float* Wq_s = (const float*)d_in[14]; const float* bq_s = (const float*)d_in[15];
  const float* Wk_s = (const float*)d_in[16]; const float* bk_s = (const float*)d_in[17];
  const float* Wv_s = (const float*)d_in[18]; const float* bv_s = (const float*)d_in[19];
  const float* Wp_v = (const float*)d_in[20]; const float* bp_v = (const float*)d_in[21];
  const float* Wp_s = (const float*)d_in[22]; const float* bp_s = (const float*)d_in[23];
  const float* g_v = (const float*)d_in[24];  const float* be_v = (const float*)d_in[25];
  const float* g_s = (const float*)d_in[26];  const float* be_s = (const float*)d_in[27];
  const float* pos_v = (const float*)d_in[28];
  const float* pos_s = (const float*)d_in[29];
  float* out = (float*)d_out;

  // ---- workspace bump allocator (256B aligned) ----
  char* wp = (char*)d_ws;
  auto alloc = [&](size_t elems, size_t esz) -> void* {
    void* r = (void*)wp;
    wp += (elems * esz + 255) & ~(size_t)255;
    return r;
  };
  const size_t MV = (size_t)B_ * LV_;   // 3136
  const size_t MS = (size_t)B_ * LS_;   // 8192
  const int LVP = 224;                  // 196 padded to 32
  const int LSP = 512;

  half_t* v16    = (half_t*)alloc(MV * FV_, 2);
  half_t* s16    = (half_t*)alloc(MS * FS_, 2);
  half_t* Wlv16  = (half_t*)alloc((size_t)DM_ * FV_, 2);
  half_t* Wls16  = (half_t*)alloc((size_t)DM_ * FS_, 2);
  half_t* Wqv16  = (half_t*)alloc((size_t)DM_ * DM_, 2);
  half_t* Wkv16  = (half_t*)alloc((size_t)DM_ * DM_, 2);
  half_t* Wvv16  = (half_t*)alloc((size_t)DM_ * DM_, 2);
  half_t* Wqs16  = (half_t*)alloc((size_t)DM_ * DM_, 2);
  half_t* Wks16  = (half_t*)alloc((size_t)DM_ * DM_, 2);
  half_t* Wvs16  = (half_t*)alloc((size_t)DM_ * DM_, 2);
  half_t* Wpv16  = (half_t*)alloc((size_t)DM_ * DM_, 2);
  half_t* Wps16  = (half_t*)alloc((size_t)DM_ * DM_, 2);
  float*  v2f    = (float*)alloc(MV * DM_, 4);
  float*  s2f    = (float*)alloc(MS * DM_, 4);
  half_t* v2h    = (half_t*)alloc(MV * DM_, 2);
  half_t* s2h    = (half_t*)alloc(MS * DM_, 2);
  half_t* qvh    = (half_t*)alloc(MV * DM_, 2);
  half_t* kvh    = (half_t*)alloc(MV * DM_, 2);
  half_t* vvh    = (half_t*)alloc(MV * DM_, 2);
  half_t* qsh    = (half_t*)alloc(MS * DM_, 2);
  half_t* ksh    = (half_t*)alloc(MS * DM_, 2);
  half_t* vsh    = (half_t*)alloc(MS * DM_, 2);
  half_t* vswzV  = (half_t*)alloc((size_t)B_ * H_ * (LVP / 32) * 4 * 32 * 16, 2);
  half_t* vswzS  = (half_t*)alloc((size_t)B_ * H_ * (LSP / 32) * 4 * 32 * 16, 2);
  half_t* ctxv16 = (half_t*)alloc(MV * DM_, 2);
  half_t* ctxs16 = (half_t*)alloc(MS * DM_, 2);
  float*  projv  = (float*)alloc(MV * DM_, 4);
  float*  projs  = (float*)alloc(MS * DM_, 4);

  auto cvt = [&](const float* x, half_t* y, size_t n) {
    unsigned blocks = (unsigned)((n + 256 * 8 - 1) / (256 * 8));
    if (blocks > 4096) blocks = 4096;
    if (blocks == 0) blocks = 1;
    cvt_f32_f16<<<blocks, 256, 0, stream>>>(x, y, n);
  };
  // ---- f32 -> f16 ----
  cvt(v, v16, MV * FV_);
  cvt(s, s16, MS * FS_);
  cvt(Wlv, Wlv16, (size_t)DM_ * FV_);
  cvt(Wls, Wls16, (size_t)DM_ * FS_);
  cvt(Wq_v, Wqv16, (size_t)DM_ * DM_);
  cvt(Wk_v, Wkv16, (size_t)DM_ * DM_);
  cvt(Wv_v, Wvv16, (size_t)DM_ * DM_);
  cvt(Wq_s, Wqs16, (size_t)DM_ * DM_);
  cvt(Wk_s, Wks16, (size_t)DM_ * DM_);
  cvt(Wv_s, Wvs16, (size_t)DM_ * DM_);
  cvt(Wp_v, Wpv16, (size_t)DM_ * DM_);
  cvt(Wp_s, Wps16, (size_t)DM_ * DM_);

  // ---- input projections (keep f32 for residual, f16 for QKV stage) ----
  gemm_f16w<<<dim3((unsigned)(MV / 64), DM_ / 64), 256, 0, stream>>>(
      v16, Wlv16, blv, v2f, v2h, (int)MV, DM_, FV_);
  gemm_f16w<<<dim3((unsigned)(MS / 64), DM_ / 64), 256, 0, stream>>>(
      s16, Wls16, bls, s2f, s2h, (int)MS, DM_, FS_);

  // ---- QKV projections (f16 outputs only) ----
  gemm_f16w<<<dim3((unsigned)(MV / 64), DM_ / 64), 256, 0, stream>>>(
      v2h, Wqv16, bq_v, nullptr, qvh, (int)MV, DM_, DM_);
  gemm_f16w<<<dim3((unsigned)(MV / 64), DM_ / 64), 256, 0, stream>>>(
      v2h, Wkv16, bk_v, nullptr, kvh, (int)MV, DM_, DM_);
  gemm_f16w<<<dim3((unsigned)(MV / 64), DM_ / 64), 256, 0, stream>>>(
      v2h, Wvv16, bv_v, nullptr, vvh, (int)MV, DM_, DM_);
  gemm_f16w<<<dim3((unsigned)(MS / 64), DM_ / 64), 256, 0, stream>>>(
      s2h, Wqs16, bq_s, nullptr, qsh, (int)MS, DM_, DM_);
  gemm_f16w<<<dim3((unsigned)(MS / 64), DM_ / 64), 256, 0, stream>>>(
      s2h, Wks16, bk_s, nullptr, ksh, (int)MS, DM_, DM_);
  gemm_f16w<<<dim3((unsigned)(MS / 64), DM_ / 64), 256, 0, stream>>>(
      s2h, Wvs16, bv_s, nullptr, vsh, (int)MS, DM_, DM_);

  // ---- V pre-swizzle into B-fragment order ----
  {
    long tot = (long)B_ * H_ * (LVP / 32) * 4 * 32;
    vswz_kernel<<<(unsigned)((tot + 255) / 256), 256, 0, stream>>>(vvh, vswzV, LV_, LVP);
    tot = (long)B_ * H_ * (LSP / 32) * 4 * 32;
    vswz_kernel<<<(unsigned)((tot + 255) / 256), 256, 0, stream>>>(vsh, vswzS, LS_, LSP);
  }

  // ---- dual cross attention ----
  const float scale = 0.125f;  // 1/sqrt(64)
  {
    const int qtV = (LV_ + 15) / 16;  // 13
    attn_kernel<<<(unsigned)(B_ * H_ * qtV), 32, 0, stream>>>(
        qvh, ksh, vswzS, ctxv16, LV_, LS_, LSP, scale);  // ctx_v = attn(q_v, k_s, v_s)
    const int qtS = LS_ / 16;         // 32
    attn_kernel<<<(unsigned)(B_ * H_ * qtS), 32, 0, stream>>>(
        qsh, kvh, vswzV, ctxs16, LS_, LV_, LVP, scale);  // ctx_s = attn(q_s, k_v, v_v)
  }

  // ---- output projections ----
  gemm_f16w<<<dim3((unsigned)(MV / 64), DM_ / 64), 256, 0, stream>>>(
      ctxv16, Wpv16, bp_v, projv, nullptr, (int)MV, DM_, DM_);
  gemm_f16w<<<dim3((unsigned)(MS / 64), DM_ / 64), 256, 0, stream>>>(
      ctxs16, Wps16, bp_s, projs, nullptr, (int)MS, DM_, DM_);

  // ---- residual + pos-emb + LayerNorm ----
  ln_fuse<<<(unsigned)(MV + MS), 256, 0, stream>>>(
      projv, projs, v2f, s2f, v_len, s_len, pos_v, pos_s,
      g_v, be_v, g_s, be_s, out);
}
